// CausalSelfAttention_72696616452423
// MI455X (gfx1250) — matrix operand order repaired and checked by reading, not compiled, not run on hardware
//
#include <hip/hip_runtime.h>

// ---------- types ----------
typedef __bf16 bf16;
typedef __attribute__((ext_vector_type(16))) __bf16 v16bf;
typedef __attribute__((ext_vector_type(4)))  __bf16 v4bf;
typedef __attribute__((ext_vector_type(2)))  __bf16 v2bf;
typedef __attribute__((ext_vector_type(8)))  float  v8f;
typedef __attribute__((ext_vector_type(4)))  float  v4f;
typedef __attribute__((ext_vector_type(4)))  unsigned int v4u;
typedef __attribute__((ext_vector_type(8)))  int    v8i;
typedef __attribute__((ext_vector_type(4)))  int    v4i;

// ---------- problem constants ----------
constexpr int B_ = 2;
constexpr int T_ = 2048;
constexpr int C_ = 1024;
constexpr int H_ = 16;
constexpr int D_ = 64;
constexpr float SCALE_ = 0.125f; // 1/sqrt(64)
constexpr size_t QKV_BYTES = (size_t)B_ * H_ * T_ * D_ * sizeof(bf16); // 8 MiB

// ---------- WMMA helpers ----------
__device__ __forceinline__ v8f wmma_bf16(v16bf a, v16bf b, v8f c) {
  // v_wmma_f32_16x16x32_bf16  (neg_a, A, neg_b, B, c_mod, C, reuse_a, reuse_b)
  return __builtin_amdgcn_wmma_f32_16x16x32_bf16(false, a, false, b, (short)0, c,
                                                 false, false);
}
// A-matrix 16x32 bf16: lane L holds row m=L%16; element j maps to
// K = j + 16*(L/16)
__device__ __forceinline__ int a_kidx(int j, int g) {
  return j + 16 * g;
}
// B-matrix 32x16 bf16: lane L holds col n=L%16; element j maps to K = j + 16*(L/16)

// 16-lane-group reductions (wave32; xor masks 1..8 stay inside each 16-lane half)
__device__ __forceinline__ float rmax16(float v) {
#pragma unroll
  for (int m = 8; m >= 1; m >>= 1) v = fmaxf(v, __shfl_xor(v, m, 32));
  return v;
}
__device__ __forceinline__ float rsum16(float v) {
#pragma unroll
  for (int m = 8; m >= 1; m >>= 1) v += __shfl_xor(v, m, 32);
  return v;
}

// ---------- Tensor Data Mover: 2D bf16 tile [rows x 64] -> LDS ----------
// D# group0: count=1 | lds_addr | global_addr(57b) | type=2
// D# group1: data_size=2B, tensor_dim0=64, tensor_dim1=T, tile_dim0=64,
//            tile_dim1=rows, tensor_dim0_stride=64
__device__ __forceinline__ void tdm_load_rows64_bf16(const bf16* gsrc,
                                                     unsigned lds_byte_off,
                                                     int rows) {
#if __has_builtin(__builtin_amdgcn_tensor_load_to_lds)
  unsigned long long ga = (unsigned long long)(size_t)gsrc;
  v4u g0;
  g0[0] = 1u;                                    // count=1, user mode
  g0[1] = lds_byte_off;                          // lds_addr
  g0[2] = (unsigned)(ga & 0xFFFFFFFFu);          // global_addr[31:0]
  g0[3] = (unsigned)((ga >> 32) & 0x01FFFFFFu)   // global_addr[56:32]
          | (2u << 30);                          // type=2 (image)
  v8i g1;
  g1[0] = (int)(1u << 16);                       // wg_mask=0, data_size=1 (2B)
  g1[1] = (int)(64u << 16);                      // tensor_dim0 = 64 (d)
  g1[2] = (int)((unsigned)T_ << 16);             // tensor_dim1 = T (keys)
  g1[3] = (int)(64u << 16);                      // tile_dim0 = 64
  g1[4] = rows;                                  // tile_dim1 = rows, tile_dim2=0
  g1[5] = 64;                                    // tensor_dim0_stride = 64
  g1[6] = 0;
  g1[7] = 0;                                     // tensor_dim1_stride = 0
  v4i z4 = {0, 0, 0, 0};
#if defined(__clang_major__) && (__clang_major__ >= 23)
  v8i z8 = {0, 0, 0, 0, 0, 0, 0, 0};
  __builtin_amdgcn_tensor_load_to_lds(g0, g1, z4, z4, z8, 0);
#else
  __builtin_amdgcn_tensor_load_to_lds(g0, g1, z4, z4, 0);
#endif
#else
  (void)gsrc; (void)lds_byte_off; (void)rows;
#endif
}

__device__ __forceinline__ void wait_tensorcnt0() {
#if __has_builtin(__builtin_amdgcn_s_wait_tensorcnt)
  __builtin_amdgcn_s_wait_tensorcnt((short)0);
#else
  asm volatile("s_wait_tensorcnt 0" ::: "memory");
#endif
  asm volatile("" ::: "memory"); // keep LDS reads below the wait
}

// =====================================================================
// Kernel 1/3: tiled GEMM  C[M,N] = A[M,K] * Bw[K,N] + bias
//   MODE 0: A = x (fp32) -> scatter bf16 Q/K/V into [B,H,T,D]
//   MODE 1: A = y (bf16) -> fp32 out [B*T, C]
// Block: 256 threads = 8 waves; tile 128(M) x 64(N) x 32(K);
// each wave computes 32x32 via 2x2 v_wmma_f32_16x16x32_bf16.
// =====================================================================
template <int MODE>
__global__ __launch_bounds__(256) void gemm_bias_kernel(
    const void* __restrict__ Aptr, const float* __restrict__ Bw,
    const float* __restrict__ bias, bf16* __restrict__ qws,
    bf16* __restrict__ kws, bf16* __restrict__ vws, float* __restrict__ outp) {
  constexpr int KK = C_;
  constexpr int NN = (MODE == 0) ? 3 * C_ : C_;
  constexpr int BM = 128, BN = 64, BK = 32, THREADS = 256;

  __shared__ bf16 As[BM][BK]; // 8 KiB
  __shared__ bf16 Bs[BK][BN]; // 4 KiB

  const int tid = threadIdx.x;
  const int lane = tid & 31;
  const int ln = lane & 15;
  const int g = lane >> 4;
  const int w = tid >> 5;      // 0..7
  const int wm = w >> 1;       // 0..3  (M sub-tile)
  const int wn = w & 1;        // 0..1  (N sub-tile)
  const int m0 = blockIdx.y * BM;
  const int n0 = blockIdx.x * BN;

  v8f acc[2][2] = {};

  for (int kk = 0; kk < KK; kk += BK) {
    // ---- stage A tile (convert fp32->bf16 if MODE 0) ----
    if constexpr (MODE == 0) {
      const float* A = (const float*)Aptr;
#pragma unroll
      for (int i = tid * 4; i < BM * BK; i += THREADS * 4) {
        int row = i >> 5, col = i & 31;
        v4f t = *(const v4f*)&A[(size_t)(m0 + row) * KK + kk + col];
        v4bf o;
        o[0] = (bf16)t[0]; o[1] = (bf16)t[1]; o[2] = (bf16)t[2]; o[3] = (bf16)t[3];
        *(v4bf*)&As[row][col] = o;
      }
    } else {
      const bf16* A = (const bf16*)Aptr;
#pragma unroll
      for (int i = tid * 4; i < BM * BK; i += THREADS * 4) {
        int row = i >> 5, col = i & 31;
        *(v4bf*)&As[row][col] =
            *(const v4bf*)&A[(size_t)(m0 + row) * KK + kk + col];
      }
    }
    // ---- stage B tile (fp32 weights -> bf16) ----
#pragma unroll
    for (int i = tid * 4; i < BK * BN; i += THREADS * 4) {
      int row = i >> 6, col = i & 63;
      v4f t = *(const v4f*)&Bw[(size_t)(kk + row) * NN + n0 + col];
      v4bf o;
      o[0] = (bf16)t[0]; o[1] = (bf16)t[1]; o[2] = (bf16)t[2]; o[3] = (bf16)t[3];
      *(v4bf*)&Bs[row][col] = o;
    }
    // speculative prefetch of next weight tile -> global_prefetch_b8
    if (kk + BK < KK)
      __builtin_prefetch(&Bw[(size_t)(kk + BK) * NN + n0 + (tid & 63)], 0, 1);
    __syncthreads();

    // ---- fragments + WMMA ----
    v16bf aF[2], bF[2];
#pragma unroll
    for (int mi = 0; mi < 2; ++mi) {
      int mrow = wm * 32 + mi * 16 + ln;
#pragma unroll
      for (int j = 0; j < 16; j += 2) {
        v2bf t = *(const v2bf*)&As[mrow][a_kidx(j, g)];
        aF[mi][j] = t[0];
        aF[mi][j + 1] = t[1];
      }
    }
#pragma unroll
    for (int ni = 0; ni < 2; ++ni) {
      int ncol = wn * 32 + ni * 16 + ln;
#pragma unroll
      for (int j = 0; j < 16; ++j) bF[ni][j] = Bs[j + 16 * g][ncol];
    }
#pragma unroll
    for (int mi = 0; mi < 2; ++mi)
#pragma unroll
      for (int ni = 0; ni < 2; ++ni)
        acc[mi][ni] = wmma_bf16(aF[mi], bF[ni], acc[mi][ni]);
    __syncthreads();
  }

  // ---- epilogue ----
#pragma unroll
  for (int mi = 0; mi < 2; ++mi) {
#pragma unroll
    for (int ni = 0; ni < 2; ++ni) {
#pragma unroll
      for (int r = 0; r < 8; ++r) {
        int m = m0 + wm * 32 + mi * 16 + r + 8 * g;
        int n = n0 + wn * 32 + ni * 16 + ln;
        float val = acc[mi][ni][r] + bias[n];
        if constexpr (MODE == 0) {
          int which = n / C_;
          int c = n % C_;
          int h = c >> 6;       // c / D_
          int d = c & 63;       // c % D_
          int bb = m >> 11;     // m / T_
          int t = m & 2047;     // m % T_
          bf16* dst = (which == 0) ? qws : (which == 1) ? kws : vws;
          dst[(((size_t)bb * H_ + h) * T_ + t) * D_ + d] = (bf16)val;
        } else {
          outp[(size_t)m * C_ + n] = val;
        }
      }
    }
  }
}

// =====================================================================
// Kernel 2: streaming (flash-style) attention with mask  k >= q.
// Block: 128 threads = 4 waves; each wave owns 16 query rows, streams
// 32-key blocks from (q0 & ~31) to T. Per block: 4 WMMA for S=Q*K^T,
// 4 WMMA for O += P*V. V tile is DMA'd into LDS by the Tensor Data
// Mover, overlapped with the S WMMAs + online softmax.
// =====================================================================
__global__ __launch_bounds__(128) void attn_kernel(const bf16* __restrict__ qws,
                                                   const bf16* __restrict__ kws,
                                                   const bf16* __restrict__ vws,
                                                   bf16* __restrict__ yws) {
  __shared__ bf16 pS[4][16][32]; // P round-trip scratch, 1 KiB/wave
  __shared__ bf16 Vs[4][32][64]; // TDM destination, 4 KiB/wave

  const int lane = threadIdx.x & 31;
  const int ln = lane & 15;
  const int g = lane >> 4;
  const int w = threadIdx.x >> 5; // wave in block: 0..3
  const int bh = blockIdx.y;      // 0..B*H-1
  const int bb = bh >> 4;         // bh / H
  const int h = bh & 15;          // bh % H
  const int q0 = (blockIdx.x * 4 + w) * 16;

  const bf16* Qp = qws + (size_t)bh * T_ * D_;
  const bf16* Kp = kws + (size_t)bh * T_ * D_;
  const bf16* Vp = vws + (size_t)bh * T_ * D_;
  const unsigned vs_off = (unsigned)(size_t)(&Vs[w][0][0]);

  // ---- load Q fragments (16 rows x 64 d) : two A-frags of 16x32 ----
  v16bf aQ[2];
#pragma unroll
  for (int ks = 0; ks < 2; ++ks) {
#pragma unroll
    for (int j = 0; j < 16; j += 2) {
      int d = ks * 32 + a_kidx(j, g);
      v2bf t = *(const v2bf*)&Qp[(size_t)(q0 + ln) * D_ + d];
      aQ[ks][j] = t[0];
      aQ[ks][j + 1] = t[1];
    }
  }

  v8f O[4] = {};            // 16x64 fp32 accumulator (4 d-segments)
  float m_run[8], l_run[8]; // per-row stats; index r -> row r + 8*g
#pragma unroll
  for (int r = 0; r < 8; ++r) {
    m_run[r] = -1e30f;
    l_run[r] = 0.0f;
  }

  const int kb0 = q0 & ~31;
  for (int kb = kb0; kb < T_; kb += 32) {
    // ---- kick off TDM: V[kb..kb+31][0..63] -> Vs[w] (overlaps S+softmax) --
    tdm_load_rows64_bf16(Vp + (size_t)kb * D_, vs_off, 32);

    // ---- B-frags of K^T: contiguous 32B loads (K^T columns are K rows) ----
    v16bf bk[2][2];
#pragma unroll
    for (int hf = 0; hf < 2; ++hf) {
      int key = kb + hf * 16 + ln;
#pragma unroll
      for (int ks = 0; ks < 2; ++ks)
        bk[hf][ks] = *(const v16bf*)&Kp[(size_t)key * D_ + ks * 32 + 16 * g];
    }
    // ---- S tiles (16 q x 32 keys), accumulate over d=64 in two steps ----
    v8f s0 = {}, s1 = {};
    s0 = wmma_bf16(aQ[0], bk[0][0], s0);
    s0 = wmma_bf16(aQ[1], bk[0][1], s0);
    s1 = wmma_bf16(aQ[0], bk[1][0], s1);
    s1 = wmma_bf16(aQ[1], bk[1][1], s1);

    // ---- online softmax (mask keeps k >= q) ----
#pragma unroll
    for (int r = 0; r < 8; ++r) {
      int rl = r + 8 * g;
      int qrow = q0 + rl;
      float v0 = s0[r] * SCALE_;
      float v1 = s1[r] * SCALE_;
      bool ok0 = (kb + ln) >= qrow;
      bool ok1 = (kb + 16 + ln) >= qrow;
      float mloc = fmaxf(ok0 ? v0 : -1e30f, ok1 ? v1 : -1e30f);
      mloc = rmax16(mloc);
      float mnew = fmaxf(m_run[r], mloc);
      float p0 = ok0 ? __expf(v0 - mnew) : 0.0f;
      float p1 = ok1 ? __expf(v1 - mnew) : 0.0f;
      float ls = rsum16(p0 + p1);
      float corr = __expf(m_run[r] - mnew);
      l_run[r] = l_run[r] * corr + ls;
      m_run[r] = mnew;
#pragma unroll
      for (int t2 = 0; t2 < 4; ++t2) O[t2][r] *= corr;
      pS[w][rl][ln] = (bf16)p0;
      pS[w][rl][16 + ln] = (bf16)p1;
    }
    // same-wave LDS producer->consumer: drain DScnt before re-reading P
    asm volatile("s_wait_dscnt 0" ::: "memory");

    // ---- A-frag of P (16 q x 32 keys) from LDS ----
    v16bf aP;
#pragma unroll
    for (int j = 0; j < 16; j += 2) {
      v2bf t = *(const v2bf*)&pS[w][ln][a_kidx(j, g)];
      aP[j] = t[0];
      aP[j + 1] = t[1];
    }

    // ---- B-frags of V (32 keys x 64 d) from the TDM'd LDS tile ----
    wait_tensorcnt0();
    v16bf bv[4];
#pragma unroll
    for (int ds = 0; ds < 4; ++ds) {
#pragma unroll
      for (int j = 0; j < 16; ++j)
        bv[ds][j] = Vs[w][j + 16 * g][ds * 16 + ln];
    }
#pragma unroll
    for (int ds = 0; ds < 4; ++ds) O[ds] = wmma_bf16(aP, bv[ds], O[ds]);
  }

  // ---- normalize + write y into [B, T, H*D] (fused transpose) ----
#pragma unroll
  for (int r = 0; r < 8; ++r) {
    float inv = 1.0f / l_run[r];
    int row = q0 + r + 8 * g;
    size_t base = ((size_t)bb * T_ + row) * C_ + (size_t)h * D_;
#pragma unroll
    for (int t2 = 0; t2 < 4; ++t2)
      yws[base + t2 * 16 + ln] = (bf16)(O[t2][r] * inv);
  }
}

// =====================================================================
extern "C" void kernel_launch(void* const* d_in, const int* in_sizes, int n_in,
                              void* d_out, int out_size, void* d_ws,
                              size_t ws_size, hipStream_t stream) {
  (void)in_sizes; (void)n_in; (void)out_size; (void)ws_size;
  const float* x    = (const float*)d_in[0];
  const float* Wqkv = (const float*)d_in[1];
  const float* bqkv = (const float*)d_in[2];
  const float* Wo   = (const float*)d_in[3];
  const float* bo   = (const float*)d_in[4];
  float* out = (float*)d_out;

  char* ws = (char*)d_ws;
  bf16* qws = (bf16*)(ws);
  bf16* kws = (bf16*)(ws + QKV_BYTES);
  bf16* vws = (bf16*)(ws + 2 * QKV_BYTES);
  bf16* yws = (bf16*)(ws + 3 * QKV_BYTES);

  // 1) QKV projection: [B*T, C] x [C, 3C] + b -> bf16 Q/K/V [B,H,T,D]
  dim3 g1((3 * C_) / 64, (B_ * T_) / 128);
  gemm_bias_kernel<0><<<g1, 256, 0, stream>>>((const void*)x, Wqkv, bqkv, qws,
                                              kws, vws, nullptr);
  // 2) attention (mask k >= q) -> bf16 y [B, T, C]
  dim3 g2(T_ / 64, B_ * H_);
  attn_kernel<<<g2, 128, 0, stream>>>(qws, kws, vws, yws);
  // 3) output projection: [B*T, C] x [C, C] + b -> fp32 out
  dim3 g3(C_ / 64, (B_ * T_) / 128);
  gemm_bias_kernel<1><<<g3, 256, 0, stream>>>((const void*)yws, Wo, bo, nullptr,
                                              nullptr, nullptr, out);
}
